// DynamicPillarVFE3D_8005819040183
// MI455X (gfx1250) — compile-verified, matrix operand order checked by reading, not run on hardware
//
#include <hip/hip_runtime.h>

// ---------------------------------------------------------------------------
// DynamicPillarVFE for MI455X (gfx1250, wave32).
// Memory-bound: recompute the cheap 11x32 GEMM per pass instead of spilling
// [N,32]/[N,64] intermediates to HBM. The dominant 64x64 GEMM runs through
// v_wmma_f32_16x16x32_f16 with w2 pre-swizzled in LDS into per-lane WMMA
// fragment layout (wide ds_load_b128 instead of scalar ds_load_u16).
// Workspace: ~41 MB (L2-resident). Assumes ws_size >= (39*V+384)*4 bytes.
// ---------------------------------------------------------------------------

#define NPTS 1500000
#define NVOX 262144
#define BN_EPS 1e-3f

// workspace layout (float offsets)
#define OFF_CNT    0                    // [V]
#define OFF_VSUM   (NVOX)               // [3V]
#define OFF_VMEAN  (4*NVOX)             // [3V]
#define OFF_XMAX   (7*NVOX)             // [32V]  (float bits, atomicMax-int)
#define OFF_STATS1 (39*NVOX)            // [64]   sum(32) | sumsq(32)
#define OFF_SC1    (39*NVOX + 64)       // [32]
#define OFF_SH1    (39*NVOX + 96)       // [32]
#define OFF_STATS2 (39*NVOX + 128)      // [128]  sum(64) | sumsq(64)
#define OFF_SC2    (39*NVOX + 256)      // [64]
#define OFF_SH2    (39*NVOX + 320)      // [64]
#define WS_FLOATS  (39*NVOX + 384)

typedef __attribute__((ext_vector_type(16))) _Float16 v16h;
typedef __attribute__((ext_vector_type(2)))  __fp16   fp16x2;  // cvt_pkrtz result type
typedef __attribute__((ext_vector_type(8)))  float    v8f;

// ---- per-point feature build + raw 11x32 GEMM (w1 staged in LDS) ----------
__device__ __forceinline__ void compute_y_raw(
    int p, const float* __restrict__ xyz, const float* __restrict__ pf,
    const int* __restrict__ inv, const float* __restrict__ vmean,
    const float* w1s, float* y)
{
  const float VS = 0.32f;
  const float P0 = -74.88f, P1 = -74.88f, P2 = -2.0f;
  float x0 = xyz[3*p+0], x1 = xyz[3*p+1], x2 = xyz[3*p+2];
  int v = inv[p];
  float fc0 = x0 - vmean[3*v+0];
  float fc1 = x1 - vmean[3*v+1];
  float fc2 = x2 - vmean[3*v+2];
  float c0 = floorf((x0 - P0) / VS);
  float c1 = floorf((x1 - P1) / VS);
  float c2 = floorf((x2 - P2) / VS);
  float ct0 = x0 - (c0 * VS + 0.5f * VS + P0);
  float ct1 = x1 - (c1 * VS + 0.5f * VS + P1);
  float ct2 = x2 - (c2 * VS + 0.5f * VS + P2);
  float dist = sqrtf(x0*x0 + x1*x1 + x2*x2);
  float feat[11] = {x0, x1, x2, pf[p], fc0, fc1, fc2, ct0, ct1, ct2, dist};
#pragma unroll
  for (int c = 0; c < 32; ++c) {
    float acc = 0.f;
#pragma unroll
    for (int k = 0; k < 11; ++k) acc = fmaf(feat[k], w1s[k*32 + c], acc);
    y[c] = acc;
  }
}

// ---- kernel 0: zero workspace + output ------------------------------------
__global__ void k_init(float* __restrict__ ws, float* __restrict__ out, int outn)
{
  long long i = (long long)blockIdx.x * blockDim.x + threadIdx.x;
  long long stride = (long long)gridDim.x * blockDim.x;
  for (long long j = i; j < WS_FLOATS; j += stride) ws[j] = 0.f;
  for (long long j = i; j < outn; j += stride) out[j] = 0.f;
}

// ---- kernel 1: scatter count + xyz sums per voxel -------------------------
__global__ void k_scatter(const float* __restrict__ xyz,
                          const int* __restrict__ inv, float* __restrict__ ws)
{
  int p = blockIdx.x * blockDim.x + threadIdx.x;
  if (p < NPTS) {
    int v = inv[p];
    atomicAdd(&ws[OFF_CNT + v], 1.f);
    atomicAdd(&ws[OFF_VSUM + 3*v + 0], xyz[3*p + 0]);
    atomicAdd(&ws[OFF_VSUM + 3*v + 1], xyz[3*p + 1]);
    atomicAdd(&ws[OFF_VSUM + 3*v + 2], xyz[3*p + 2]);
  }
}

// ---- kernel 2: vmean = vsum / max(cnt, 1) ---------------------------------
__global__ void k_vmean(float* __restrict__ ws)
{
  int v = blockIdx.x * blockDim.x + threadIdx.x;
  if (v < NVOX) {
    float inv_c = 1.f / fmaxf(ws[OFF_CNT + v], 1.f);
    ws[OFF_VMEAN + 3*v + 0] = ws[OFF_VSUM + 3*v + 0] * inv_c;
    ws[OFF_VMEAN + 3*v + 1] = ws[OFF_VSUM + 3*v + 1] * inv_c;
    ws[OFF_VMEAN + 3*v + 2] = ws[OFF_VSUM + 3*v + 2] * inv_c;
  }
}

// ---- kernel 3: PFN1 batch-norm statistics ---------------------------------
__global__ void k_pfn1_stats(const float* __restrict__ xyz, const float* __restrict__ pf,
                             const int* __restrict__ inv, const float* __restrict__ w1,
                             float* __restrict__ ws)
{
  __shared__ float w1s[352];
  __shared__ float ls[64];
  int tid = threadIdx.x;
  for (int i = tid; i < 352; i += blockDim.x) w1s[i] = w1[i];
  if (tid < 64) ls[tid] = 0.f;
  __syncthreads();
  int p = blockIdx.x * blockDim.x + tid;
  if (p < NPTS) {
    float y[32];
    compute_y_raw(p, xyz, pf, inv, ws + OFF_VMEAN, w1s, y);
#pragma unroll
    for (int c = 0; c < 32; ++c) {
      atomicAdd(&ls[c], y[c]);
      atomicAdd(&ls[32 + c], y[c] * y[c]);
    }
  }
  __syncthreads();
  if (tid < 64) atomicAdd(&ws[OFF_STATS1 + tid], ls[tid]);
}

// ---- BN finalize: fold gamma/beta into scale/shift ------------------------
__global__ void k_bn_finalize(const float* __restrict__ stats,
                              const float* __restrict__ g, const float* __restrict__ b,
                              float* __restrict__ sc, float* __restrict__ sh,
                              int C, float invN)
{
  int c = threadIdx.x;
  if (c < C) {
    float mu  = stats[c] * invN;
    float var = stats[C + c] * invN - mu * mu;
    float s = g[c] * rsqrtf(var + BN_EPS);
    sc[c] = s;
    sh[c] = b[c] - mu * s;
  }
}

// ---- kernel 4: PFN1 apply + segment-max into x_max ------------------------
__global__ void k_pfn1_apply(const float* __restrict__ xyz, const float* __restrict__ pf,
                             const int* __restrict__ inv, const float* __restrict__ w1,
                             float* __restrict__ ws)
{
  __shared__ float w1s[352];
  __shared__ float sc[32], sh[32];
  int tid = threadIdx.x;
  for (int i = tid; i < 352; i += blockDim.x) w1s[i] = w1[i];
  if (tid < 32) { sc[tid] = ws[OFF_SC1 + tid]; sh[tid] = ws[OFF_SH1 + tid]; }
  __syncthreads();
  int p = blockIdx.x * blockDim.x + tid;
  if (p < NPTS) {
    float y[32];
    compute_y_raw(p, xyz, pf, inv, ws + OFF_VMEAN, w1s, y);
    int v = inv[p];
    int* xm = (int*)(ws + OFF_XMAX) + v * 32;
#pragma unroll
    for (int c = 0; c < 32; ++c) {
      float t = fmaxf(fmaf(y[c], sc[c], sh[c]), 0.f);
      atomicMax(&xm[c], __float_as_int(t));   // valid: t >= 0, buffer init 0
    }
  }
}

// ---- stage w2 into LDS, pre-swizzled into per-lane WMMA B-fragments -------
// Fragment t = k*4 + n (k = K-half of x, n = 16-col output tile). For lane l,
// v16h element e maps to w2 row (k*32 + r), col (n*16 + (l&15)) with
// r = (e<8) ? (l>>4)*8+e : 8+(l>>4)*8+e  (ISA 16-bit operand striping).
// Stored contiguously so each fragment is one aligned 32-byte LDS load.
__device__ __forceinline__ void stage_w2_swizzled(
    const float* __restrict__ w2, _Float16* w2sw, int tid)
{
  for (int i = tid; i < 4096; i += 256) {
    int e    = i & 15;
    int lane = (i >> 4) & 31;
    int t    = i >> 9;
    int k = t >> 2, n = t & 3;
    int half = lane >> 4, m = lane & 15;
    int r = (e < 8) ? (half * 8 + e) : (8 + half * 8 + e);
    w2sw[i] = (_Float16)w2[(k * 32 + r) * 64 + (n * 16 + m)];
  }
}

// ---- shared WMMA tile: z[16x64] = [y_bn | x_max] @ w2 ---------------------
// Lanes l and l+16 redundantly compute row M = l&15 and keep complementary
// K-halves per the ISA A-matrix striping. A is packed with v_cvt_pk_rtz.
__device__ __forceinline__ void pfn2_tile(
    int base, bool tval, int lane,
    const float* __restrict__ xyz, const float* __restrict__ pf,
    const int* __restrict__ inv, const float* __restrict__ ws,
    const float* w1s, const float* sc1, const float* sh1,
    const v16h* w2v, v8f acc[4])
{
  const int m = lane & 15;
  const int half = lane >> 4;
  const int p = tval ? base + m : 0;

  float y[32];
  compute_y_raw(p, xyz, pf, inv, ws + OFF_VMEAN, w1s, y);
  const int v = inv[p];
  const float* xm = ws + OFF_XMAX + (size_t)v * 32;
  const float mask = tval ? 1.f : 0.f;

  union Frag { v16h v; fp16x2 p[8]; };
  Frag a0, a1;
#pragma unroll
  for (int j = 0; j < 8; ++j) {
    int e0 = 2 * j, e1 = 2 * j + 1;                 // same K-region (split at 8)
    int c0 = (e0 < 8) ? (half * 8 + e0) : (8 + half * 8 + e0);
    int c1 = (e1 < 8) ? (half * 8 + e1) : (8 + half * 8 + e1);
    float y0 = fmaxf(fmaf(y[c0], sc1[c0], sh1[c0]), 0.f) * mask;
    float y1 = fmaxf(fmaf(y[c1], sc1[c1], sh1[c1]), 0.f) * mask;
    a0.p[j] = __builtin_amdgcn_cvt_pkrtz(y0, y1);
    a1.p[j] = __builtin_amdgcn_cvt_pkrtz(xm[c0] * mask, xm[c1] * mask);
  }

#pragma unroll
  for (int n = 0; n < 4; ++n) {
    v16h B0 = w2v[n * 32 + lane];          // K = 0..31  fragment
    v16h B1 = w2v[(4 + n) * 32 + lane];    // K = 32..63 fragment
    v8f c8 = acc[n];
    c8 = __builtin_amdgcn_wmma_f32_16x16x32_f16(false, a0.v, false, B0,
                                                (short)0, c8, false, false);
    c8 = __builtin_amdgcn_wmma_f32_16x16x32_f16(false, a1.v, false, B1,
                                                (short)0, c8, false, false);
    acc[n] = c8;
  }
}

// ---- kernel 5: PFN2 batch-norm statistics (WMMA) --------------------------
__global__ void __launch_bounds__(256)
k_pfn2_stats(const float* __restrict__ xyz, const float* __restrict__ pf,
             const int* __restrict__ inv, const float* __restrict__ w1,
             const float* __restrict__ w2, float* __restrict__ ws)
{
  __shared__ float w1s[352];
  __shared__ float sc1[32], sh1[32];
  __shared__ __attribute__((aligned(32))) _Float16 w2sw[4096];
  __shared__ float ls[128];
  int tid = threadIdx.x;
  for (int i = tid; i < 352; i += 256) w1s[i] = w1[i];
  stage_w2_swizzled(w2, w2sw, tid);
  if (tid < 32) { sc1[tid] = ws[OFF_SC1 + tid]; sh1[tid] = ws[OFF_SH1 + tid]; }
  if (tid < 128) ls[tid] = 0.f;
  __syncthreads();

  int wave = blockIdx.x * 8 + (tid >> 5);
  int lane = tid & 31;
  int base = wave * 16;
  bool tval = base < NPTS;

  v8f acc[4] = {};
  pfn2_tile(base, tval, lane, xyz, pf, inv, ws, w1s, sc1, sh1,
            (const v16h*)w2sw, acc);

#pragma unroll
  for (int n = 0; n < 4; ++n) {
    float s = 0.f, q = 0.f;
#pragma unroll
    for (int r = 0; r < 8; ++r) { float z = acc[n][r]; s += z; q += z * z; }
    int ch = n * 16 + (lane & 15);
    atomicAdd(&ls[ch], s);
    atomicAdd(&ls[64 + ch], q);
  }
  __syncthreads();
  if (tid < 128) atomicAdd(&ws[OFF_STATS2 + tid], ls[tid]);
}

// ---- kernel 6: PFN2 apply (WMMA) + segment-max into output ----------------
__global__ void __launch_bounds__(256)
k_pfn2_apply(const float* __restrict__ xyz, const float* __restrict__ pf,
             const int* __restrict__ inv, const float* __restrict__ w1,
             const float* __restrict__ w2, float* __restrict__ ws,
             float* __restrict__ out)
{
  __shared__ float w1s[352];
  __shared__ float sc1[32], sh1[32];
  __shared__ float sc2[64], sh2[64];
  __shared__ __attribute__((aligned(32))) _Float16 w2sw[4096];
  int tid = threadIdx.x;
  for (int i = tid; i < 352; i += 256) w1s[i] = w1[i];
  stage_w2_swizzled(w2, w2sw, tid);
  if (tid < 32) { sc1[tid] = ws[OFF_SC1 + tid]; sh1[tid] = ws[OFF_SH1 + tid]; }
  if (tid < 64) { sc2[tid] = ws[OFF_SC2 + tid]; sh2[tid] = ws[OFF_SH2 + tid]; }
  __syncthreads();

  int wave = blockIdx.x * 8 + (tid >> 5);
  int lane = tid & 31;
  int half = lane >> 4;
  int base = wave * 16;
  bool tval = base < NPTS;

  v8f acc[4] = {};
  pfn2_tile(base, tval, lane, xyz, pf, inv, ws, w1s, sc1, sh1,
            (const v16h*)w2sw, acc);

  if (tval) {
#pragma unroll
    for (int n = 0; n < 4; ++n) {
      int ch = n * 16 + (lane & 15);
      float sc = sc2[ch], sh = sh2[ch];
#pragma unroll
      for (int r = 0; r < 8; ++r) {
        int M = half * 8 + r;               // C-matrix row per ISA layout
        int pp = base + M;
        float val = fmaxf(fmaf(acc[n][r], sc, sh), 0.f);
        int vox = inv[pp];
        atomicMax((int*)out + (size_t)vox * 64 + ch, __float_as_int(val));
      }
    }
  }
}

// ---------------------------------------------------------------------------
extern "C" void kernel_launch(void* const* d_in, const int* in_sizes, int n_in,
                              void* d_out, int out_size, void* d_ws, size_t ws_size,
                              hipStream_t stream)
{
  const float* xyz = (const float*)d_in[0];
  const float* pf  = (const float*)d_in[1];
  const float* w1  = (const float*)d_in[2];
  const float* g1  = (const float*)d_in[3];
  const float* b1  = (const float*)d_in[4];
  const float* w2  = (const float*)d_in[5];
  const float* g2  = (const float*)d_in[6];
  const float* b2  = (const float*)d_in[7];
  const int*   inv = (const int*)d_in[8];
  float* ws  = (float*)d_ws;
  float* out = (float*)d_out;

  const int ptBlocks  = (NPTS + 255) / 256;
  const int voxBlocks = (NVOX + 255) / 256;
  const int nWaves    = NPTS / 16;            // 93750 tiles, exact
  const int wmBlocks  = (nWaves + 7) / 8;     // 8 waves (256 threads) / block
  const float invN = 1.f / (float)NPTS;

  k_init<<<2048, 256, 0, stream>>>(ws, out, out_size);
  k_scatter<<<ptBlocks, 256, 0, stream>>>(xyz, inv, ws);
  k_vmean<<<voxBlocks, 256, 0, stream>>>(ws);
  k_pfn1_stats<<<ptBlocks, 256, 0, stream>>>(xyz, pf, inv, w1, ws);
  k_bn_finalize<<<1, 64, 0, stream>>>(ws + OFF_STATS1, g1, b1,
                                      ws + OFF_SC1, ws + OFF_SH1, 32, invN);
  k_pfn1_apply<<<ptBlocks, 256, 0, stream>>>(xyz, pf, inv, w1, ws);
  k_pfn2_stats<<<wmBlocks, 256, 0, stream>>>(xyz, pf, inv, w1, w2, ws);
  k_bn_finalize<<<1, 64, 0, stream>>>(ws + OFF_STATS2, g2, b2,
                                      ws + OFF_SC2, ws + OFF_SH2, 64, invN);
  k_pfn2_apply<<<wmBlocks, 256, 0, stream>>>(xyz, pf, inv, w1, w2, ws, out);
}
